// MultiHeadAttention_37752762532255
// MI455X (gfx1250) — matrix operand order repaired and checked by reading, not compiled, not run on hardware
//
#include <hip/hip_runtime.h>

// ---------------------------------------------------------------------------
// MHA forward for B=2, S=2048, D_HIDN=1024, N_HEAD=16, D_HEAD=64 on gfx1250.
// Strategy: bf16 WMMA (v_wmma_f32_16x16x32_bf16, f32 accum), f32 softmax.
// attn_prob (537MB) is a required output -> staged pipeline, not flash-attn.
// All WMMA operands are pre-packed into fragment-major bf16 buffers matching
// the CDNA5 ISA 16x32 A / 32x16 B lane layouts so GEMM loads are b128 vectors.
// Workspace usage: ~50 MB (regions reused across stages).
// ---------------------------------------------------------------------------

typedef __attribute__((ext_vector_type(16))) __bf16 bf16x16;
typedef __attribute__((ext_vector_type(8)))  float  fx8;

#define NB   2
#define NS   2048
#define NH   16
#define DH   64
#define DM   1024          // N_HEAD * D_HEAD == D_HIDN

// float -> bf16, round-to-nearest-even (bit-level, avoids relying on __bf16 casts)
static __device__ __forceinline__ unsigned short f2bf(float f) {
    unsigned int u = __float_as_uint(f);
    u += 0x7FFFu + ((u >> 16) & 1u);
    return (unsigned short)(u >> 16);
}

union BF16Frag { bf16x16 v; unsigned short u[16]; };

// ---------------------------------------------------------------------------
// Pack fp32 row-major [M,K] (leading dim ld) into bf16 A-fragment layout:
//   fragment (mt,kt) covers rows mt*16..+15, cols kt*32..+31
//   per lane: row = lane&15, khalf = lane>>4
//   element i: k = kt*32 + (i&7) + 8*khalf + 16*(i>>3)     (ISA 7.12.2)
// Output: [frag][lane][16] contiguous (coalesced writes).
// ---------------------------------------------------------------------------
__global__ __launch_bounds__(256) void pack_a_kernel(
    const float* __restrict__ in, unsigned short* __restrict__ out,
    int M, int K, int ld)
{
    long long idx = (long long)blockIdx.x * 256 + threadIdx.x;
    if (idx >= (long long)M * K) return;
    int i    = (int)(idx & 15);
    int lane = (int)((idx >> 4) & 31);
    long long frag = idx >> 9;
    int Kt = K >> 5;
    int mt = (int)(frag / Kt);
    int kt = (int)(frag - (long long)mt * Kt);
    int row = lane & 15;
    int kh  = (lane >> 4) & 1;
    int kk  = (i & 7) + (kh << 3) + ((i >> 3) << 4);
    out[idx] = f2bf(in[(size_t)(mt * 16 + row) * ld + kt * 32 + kk]);
}

// ---------------------------------------------------------------------------
// Pack fp32 matrix element (k,n) = in[k*ld_k + n*ld_n] into bf16 B-fragment
// layout (32x16 tiles):
//   per lane: n = nt*16 + (lane&15), khalf = lane>>4
//   element i: k = kt*32 + (i&7) + 8*khalf + 16*(i>>3)     (ISA 7.12.2)
// Batched via blockIdx.y (strides in elements).
// ---------------------------------------------------------------------------
__global__ __launch_bounds__(256) void pack_b_kernel(
    const float* __restrict__ in, unsigned short* __restrict__ out,
    int K, int N, long long ld_k, long long ld_n,
    long long strideIn, long long strideOut)
{
    long long idx = (long long)blockIdx.x * 256 + threadIdx.x;
    if (idx >= (long long)K * N) return;
    int i    = (int)(idx & 15);
    int lane = (int)((idx >> 4) & 31);
    long long frag = idx >> 9;
    int Nt = N >> 4;
    int kt = (int)(frag / Nt);
    int nt = (int)(frag - (long long)kt * Nt);
    int n  = nt * 16 + (lane & 15);
    int k  = kt * 32 + (i & 7) + ((lane >> 4) << 3) + ((i >> 3) << 4);
    out[(long long)blockIdx.y * strideOut + idx] =
        f2bf(in[(long long)blockIdx.y * strideIn + (long long)k * ld_k + (long long)n * ld_n]);
}

#define WMMA_BF16(c, a, b) \
    (c) = __builtin_amdgcn_wmma_f32_16x16x32_bf16(false, (a), false, (b), (short)0, (c), false, false)

// ---------------------------------------------------------------------------
// Generic batched bf16 WMMA GEMM: C = alpha * (A @ B) + bias
// A: fragment-packed [M/16][K/32] ; B: fragment-packed [K/32][N/16]
// Each wave computes a 16x64 strip (1 A frag x 4 B frags per K step).
// mode 0: C row-major [M,N] (ldc = N). mode 1: head-split scatter:
//   m = b*2048+s, n = h*64+d -> C[((b*16+h)*2048+s)*64+d]
// ---------------------------------------------------------------------------
__global__ __launch_bounds__(256) void gemm_bf16_wmma(
    const unsigned short* __restrict__ Ap, const unsigned short* __restrict__ Bp,
    float* __restrict__ C, const float* __restrict__ bias,
    int M, int N, int K, float alpha, int mode,
    long long batchA, long long batchB, long long batchC)
{
    int lane = threadIdx.x & 31;
    int wave = threadIdx.x >> 5;
    int tilesN = N >> 6;                       // strips of 64 along N
    int totalTiles = (M >> 4) * tilesN;
    int gw = blockIdx.x * 8 + wave;            // wave-uniform
    if (gw >= totalTiles) return;              // whole-wave exit: EXEC stays all-1
    int mt  = gw / tilesN;
    int nt4 = gw - mt * tilesN;

    const unsigned short* A = Ap + (long long)blockIdx.y * batchA;
    const unsigned short* B = Bp + (long long)blockIdx.y * batchB;
    float* Cb = C + (long long)blockIdx.y * batchC;

    int Kt  = K >> 5;
    int NtT = N >> 4;

    fx8 c0 = {}, c1 = {}, c2 = {}, c3 = {};
    for (int kt = 0; kt < Kt; ++kt) {
        bf16x16 a  = *(const bf16x16*)(A + (((long long)mt * Kt + kt) << 9) + (lane << 4));
        const unsigned short* brow = B + (((long long)kt * NtT + nt4 * 4) << 9) + (lane << 4);
        bf16x16 b0 = *(const bf16x16*)(brow);
        bf16x16 b1 = *(const bf16x16*)(brow + 512);
        bf16x16 b2 = *(const bf16x16*)(brow + 1024);
        bf16x16 b3 = *(const bf16x16*)(brow + 1536);
        WMMA_BF16(c0, a, b0);
        WMMA_BF16(c1, a, b1);
        WMMA_BF16(c2, a, b2);
        WMMA_BF16(c3, a, b3);
    }

    int nlo = lane & 15;
    int mhi = (lane >> 4) << 3;
#pragma unroll
    for (int r = 0; r < 8; ++r) {
        int m = mt * 16 + mhi + r;
        float vals[4] = { c0[r], c1[r], c2[r], c3[r] };
#pragma unroll
        for (int j = 0; j < 4; ++j) {
            int n = nt4 * 64 + j * 16 + nlo;
            float v = vals[j] * alpha + (bias ? bias[n] : 0.0f);
            if (mode == 0) {
                Cb[(long long)m * N + n] = v;
            } else {
                int b = m >> 11, s = m & 2047;
                int h = n >> 6,  d = n & 63;
                Cb[(((long long)(b * NH + h) * NS + s) << 6) + d] = v;
            }
        }
    }
}

// ---------------------------------------------------------------------------
// Row softmax with mask over attn region [B*H*S rows, S cols], in place.
// mask is [B,S,S] bool (1 byte), independent of h (reference semantics).
// ---------------------------------------------------------------------------
__global__ __launch_bounds__(256) void softmax_mask_kernel(
    float* __restrict__ attn, const unsigned char* __restrict__ mask)
{
    __shared__ float red[256];
    int row = blockIdx.x;                         // 0 .. B*H*S-1
    int tid = threadIdx.x;
    float* p = attn + (long long)row * NS;
    int b  = row >> 15;                           // row / (H*S), H*S = 32768
    int sq = row & (NS - 1);
    const unsigned char* mrow = mask + ((long long)b * NS + sq) * NS;

    float vals[8];
    float mx = -3.4e38f;
#pragma unroll
    for (int it = 0; it < 8; ++it) {
        int c = tid + it * 256;
        float x = p[c];
        if (mrow[c]) x = -1e9f;
        vals[it] = x;
        mx = fmaxf(mx, x);
    }
    red[tid] = mx; __syncthreads();
    for (int s = 128; s > 0; s >>= 1) {
        if (tid < s) red[tid] = fmaxf(red[tid], red[tid + s]);
        __syncthreads();
    }
    mx = red[0]; __syncthreads();

    float sum = 0.0f;
#pragma unroll
    for (int it = 0; it < 8; ++it) {
        float e = __expf(vals[it] - mx);
        vals[it] = e;
        sum += e;
    }
    red[tid] = sum; __syncthreads();
    for (int s = 128; s > 0; s >>= 1) {
        if (tid < s) red[tid] += red[tid + s];
        __syncthreads();
    }
    float inv = 1.0f / red[0];
#pragma unroll
    for (int it = 0; it < 8; ++it)
        p[tid + it * 256] = vals[it] * inv;
}

// ---------------------------------------------------------------------------
// Context GEMM: per (b,h): ctx[S,64] = P[S,S] @ V[S,64].
// A (probs) is fp32 in d_out; converted to bf16 A-fragments on the fly:
// per lane the A fragment's 16 K-indices are two runs of 8 consecutive cols
// (kk = 8*kh + 0..7 and 16 + 8*kh + 0..7) -> two float4-pair loads.
// Output written fp32 in [B, S, H*64] layout (the transpose+reshape).
// ---------------------------------------------------------------------------
__global__ __launch_bounds__(256) void context_wmma_kernel(
    const float* __restrict__ P, const unsigned short* __restrict__ Vp,
    float* __restrict__ ctx)
{
    int lane = threadIdx.x & 31;
    int wave = threadIdx.x >> 5;
    int bh = blockIdx.y;                               // 0..31
    int mt = blockIdx.x * 8 + wave;                    // 0..127
    const float* Pb = P + (long long)bh * NS * NS;
    const unsigned short* Vb = Vp + (long long)bh * (NS * DH);

    int row = mt * 16 + (lane & 15);
    int kh  = lane >> 4;
    const float* prow = Pb + (long long)row * NS + (kh << 3);

    fx8 c0 = {}, c1 = {}, c2 = {}, c3 = {};
    for (int kt = 0; kt < NS / 32; ++kt) {
        const float4* q4 = (const float4*)(prow + kt * 32);
        float4 a0 = q4[0], a1 = q4[1], a2 = q4[4], a3 = q4[5];
        BF16Frag af;
        af.u[0]  = f2bf(a0.x); af.u[1]  = f2bf(a0.y); af.u[2]  = f2bf(a0.z); af.u[3]  = f2bf(a0.w);
        af.u[4]  = f2bf(a1.x); af.u[5]  = f2bf(a1.y); af.u[6]  = f2bf(a1.z); af.u[7]  = f2bf(a1.w);
        af.u[8]  = f2bf(a2.x); af.u[9]  = f2bf(a2.y); af.u[10] = f2bf(a2.z); af.u[11] = f2bf(a2.w);
        af.u[12] = f2bf(a3.x); af.u[13] = f2bf(a3.y); af.u[14] = f2bf(a3.z); af.u[15] = f2bf(a3.w);

        const unsigned short* brow = Vb + (((long long)kt * 4) << 9) + (lane << 4);
        bf16x16 b0 = *(const bf16x16*)(brow);
        bf16x16 b1 = *(const bf16x16*)(brow + 512);
        bf16x16 b2 = *(const bf16x16*)(brow + 1024);
        bf16x16 b3 = *(const bf16x16*)(brow + 1536);
        WMMA_BF16(c0, af.v, b0);
        WMMA_BF16(c1, af.v, b1);
        WMMA_BF16(c2, af.v, b2);
        WMMA_BF16(c3, af.v, b3);
    }

    int b = bh >> 4, h = bh & 15;
    int nlo = lane & 15;
    int mhi = kh << 3;
#pragma unroll
    for (int r = 0; r < 8; ++r) {
        int m = mt * 16 + mhi + r;
        long long base = ((long long)b * NS + m) * DM + h * DH;
        float vals[4] = { c0[r], c1[r], c2[r], c3[r] };
#pragma unroll
        for (int j = 0; j < 4; ++j)
            ctx[base + j * 16 + nlo] = vals[j];
    }
}

// ---------------------------------------------------------------------------
// Host-side orchestration (all launches on `stream`, stream-ordered deps).
// ---------------------------------------------------------------------------
extern "C" void kernel_launch(void* const* d_in, const int* in_sizes, int n_in,
                              void* d_out, int out_size, void* d_ws, size_t ws_size,
                              hipStream_t stream) {
    const float* qin  = (const float*)d_in[0];
    const float* kin  = (const float*)d_in[1];
    const float* vin  = (const float*)d_in[2];
    const unsigned char* mask = (const unsigned char*)d_in[3];
    const float* w_q = (const float*)d_in[4];
    const float* b_q = (const float*)d_in[5];
    const float* w_k = (const float*)d_in[6];
    const float* b_k = (const float*)d_in[7];
    const float* w_v = (const float*)d_in[8];
    const float* b_v = (const float*)d_in[9];
    const float* w_o = (const float*)d_in[10];
    const float* b_o = (const float*)d_in[11];

    float* out  = (float*)d_out;                               // [B,S,1024]
    float* attn = out + (long long)NB * NS * DM;               // [B,H,S,S]

    // workspace regions (bytes), 256-aligned; total ~50 MB
    char* ws = (char*)d_ws;
    unsigned short* ACT = (unsigned short*)(ws + 0);           // 8 MB  (packed A, 4096x1024)
    unsigned short* WB  = (unsigned short*)(ws + 8388608);     // 2 MB  (packed weight B)
    float*          HF  = (float*)(ws + 10485760);             // 16 MB (head-split fp32 / ctx fp32)
    unsigned short* QP  = (unsigned short*)(ws + 27262976);    // 8 MB  (Q A-packed per head)
    unsigned short* KTP = (unsigned short*)(ws + 35651584);    // 8 MB  (K^T B-packed per head)
    unsigned short* VP  = (unsigned short*)(ws + 44040192);    // 8 MB  (V B-packed per head)

    const int M  = NB * NS;        // 4096
    const long long BHS = (long long)NB * NH * NS;             // 65536

    // ---- Q projection chain ----
    pack_a_kernel<<<16384, 256, 0, stream>>>(qin, ACT, M, DM, DM);
    pack_b_kernel<<<dim3(4096, 1), 256, 0, stream>>>(w_q, WB, DM, DM, DM, 1, 0, 0);
    gemm_bf16_wmma<<<dim3(512, 1), 256, 0, stream>>>(ACT, WB, HF, b_q, M, DM, DM, 1.0f, 1, 0, 0, 0);
    pack_a_kernel<<<16384, 256, 0, stream>>>(HF, QP, (int)BHS, DH, DH);

    // ---- K projection chain (K^T is the B matrix of the scores GEMM) ----
    pack_a_kernel<<<16384, 256, 0, stream>>>(kin, ACT, M, DM, DM);
    pack_b_kernel<<<dim3(4096, 1), 256, 0, stream>>>(w_k, WB, DM, DM, DM, 1, 0, 0);
    gemm_bf16_wmma<<<dim3(512, 1), 256, 0, stream>>>(ACT, WB, HF, b_k, M, DM, DM, 1.0f, 1, 0, 0, 0);
    // element (k=d, n=s) = Kh[s*64+d]  -> ld_k=1, ld_n=64 ; batch = one (b,h)
    pack_b_kernel<<<dim3(512, 32), 256, 0, stream>>>(HF, KTP, DH, NS, 1, DH, NS * DH, NS * DH);

    // ---- V projection chain ----
    pack_a_kernel<<<16384, 256, 0, stream>>>(vin, ACT, M, DM, DM);
    pack_b_kernel<<<dim3(4096, 1), 256, 0, stream>>>(w_v, WB, DM, DM, DM, 1, 0, 0);
    gemm_bf16_wmma<<<dim3(512, 1), 256, 0, stream>>>(ACT, WB, HF, b_v, M, DM, DM, 1.0f, 1, 0, 0, 0);
    pack_b_kernel<<<dim3(512, 32), 256, 0, stream>>>(HF, VP, NS, DH, DH, 1, NS * DH, NS * DH);

    // ---- scores = (Q @ K^T) * 1/sqrt(64), batched over 32 (b,h) ----
    gemm_bf16_wmma<<<dim3(512, 32), 256, 0, stream>>>(QP, KTP, attn, nullptr,
        NS, NS, DH, 0.125f, 0, NS * DH, NS * DH, (long long)NS * NS);

    // ---- masked softmax in place on attn ----
    softmax_mask_kernel<<<(unsigned)BHS, 256, 0, stream>>>(attn, mask);

    // ---- context = P @ V (fp32 probs -> bf16 on the fly), writes [B,S,1024] ----
    context_wmma_kernel<<<dim3(16, 32), 256, 0, stream>>>(attn, VP, HF);

    // ---- output projection: out = ctx @ w_o + b_o ----
    pack_a_kernel<<<16384, 256, 0, stream>>>(HF, ACT, M, DM, DM);
    pack_b_kernel<<<dim3(4096, 1), 256, 0, stream>>>(w_o, WB, DM, DM, DM, 1, 0, 0);
    gemm_bf16_wmma<<<dim3(512, 1), 256, 0, stream>>>(ACT, WB, out, b_o, M, DM, DM, 1.0f, 0, 0, 0, 0);
}
